// MultiHeadAttentionLayer_34342558499464
// MI455X (gfx1250) — matrix operand order repaired and checked by reading, not compiled, not run on hardware
//
#include <hip/hip_runtime.h>

// ---- problem constants ----
constexpr int N_TOK = 32768;
constexpr int EMB   = 1024;
constexpr int NH    = 16;
constexpr int HD    = 64;
constexpr int Q3    = 3 * EMB;   // 3072
constexpr int TM    = 32;        // tokens per block (2 WMMA row-tiles)
constexpr int NTHR  = 512;       // 16 wave32 per block

typedef __attribute__((ext_vector_type(16))) __bf16 v16bf;
typedef __attribute__((ext_vector_type(8)))  __bf16 v8bf;
typedef __attribute__((ext_vector_type(4)))  __bf16 v4bf;
typedef __attribute__((ext_vector_type(8)))  float  v8f;

union V16U { v16bf v; v8bf h[2]; };

// ---- kernel 0: convert W (3072x1024 f32) -> bf16 in workspace ----
__global__ __launch_bounds__(256)
void cvt_w_bf16(const float* __restrict__ W, __bf16* __restrict__ Wb) {
    int i = (blockIdx.x * 256 + threadIdx.x) * 4;
    float4 w = *(const float4*)(W + i);
    v4bf o;
    o[0] = (__bf16)w.x; o[1] = (__bf16)w.y;
    o[2] = (__bf16)w.z; o[3] = (__bf16)w.w;
    *(v4bf*)(Wb + i) = o;
}

// ---- fused kernel: qkv projection + per-token head attention ----
// block = 512 threads (16 wave32), each block handles 32 tokens.
// Each wave keeps TWO f32 accumulators (row-tiles) so every B fragment
// fetched from L2 feeds two v_wmma ops. The k-loop is tiled 4x8 with the
// outer k-block loop kept rolled, so A fragments stay short-lived LDS
// loads (prevents LICM from hoisting 512 VGPRs of A and spilling).
// LDS: [0,64K)     h tile bf16  (32 x 1024)
//      [64K,256K)  qkv bf16     (32 x 3072)
//      [256K,272K) per-wave attn staging (16 x 16 x 32 bf16, K padded)
__global__ __launch_bounds__(NTHR)
void mha_fused(const float* __restrict__ hptr,
               const __bf16* __restrict__ Wb,
               const float* __restrict__ bias,
               float* __restrict__ out)
{
    extern __shared__ char smem_raw[];
    __bf16* sh_h   = (__bf16*)smem_raw;                        // 32*1024
    __bf16* sh_qkv = (__bf16*)(smem_raw + 65536);              // 32*3072
    __bf16* sh_at  = (__bf16*)(smem_raw + 65536 + 196608);     // 16*16*32

    const int tid  = threadIdx.x;
    const int wave = tid >> 5;
    const int lane = tid & 31;
    const int half = lane >> 4;     // which 16-lane half of the wave
    const int l16  = lane & 15;
    const int tok0 = blockIdx.x * TM;

    // ---- stage 0: load + convert h tile to bf16 LDS (float4 -> 4x bf16) ----
    #pragma unroll
    for (int it = 0; it < (TM * EMB) / (NTHR * 4); ++it) {
        int idx = (it * NTHR + tid) * 4;
        float4 w = *(const float4*)(hptr + (size_t)(tok0 + (idx >> 10)) * EMB + (idx & 1023));
        v4bf o;
        o[0] = (__bf16)w.x; o[1] = (__bf16)w.y;
        o[2] = (__bf16)w.z; o[3] = (__bf16)w.w;
        *(v4bf*)(sh_h + idx) = o;
    }
    __syncthreads();

    // ---- phase 1: qkv = h @ W^T + b, 192 feature-tiles x 2 row-tiles ----
    #pragma clang loop unroll(disable)
    for (int i = 0; i < 12; ++i) {
        const int nt = wave + (i << 4);          // 0..191, interleaved per wave
        const int featBase = nt << 4;
        v8f acc0 = {};
        v8f acc1 = {};
        #pragma clang loop unroll(disable)
        for (int kb = 0; kb < 4; ++kb) {         // rolled: keeps A loads in-loop
            #pragma unroll
            for (int ku = 0; ku < 8; ++ku) {
                const int ks = (kb << 3) + ku;
                // A row-tiles (tokens 0..15 / 16..31), contiguous k order, same as B:
                // row = l16, half0: K 0..15, half1: K 16..31
                V16U a0, a1;
                const __bf16* ap0 = &sh_h[l16 * EMB + (ks << 5) + (half << 4)];
                a0.h[0] = *(const v8bf*)(ap0);
                a0.h[1] = *(const v8bf*)(ap0 + 8);
                const __bf16* ap1 = ap0 + 16 * EMB;
                a1.h[0] = *(const v8bf*)(ap1);
                a1.h[1] = *(const v8bf*)(ap1 + 8);
                // B (32 K x 16 feats): column = featBase+l16,
                // half0 holds K 0..15, half1 holds K 16..31 (contiguous 32B)
                const __bf16* bp = &Wb[(size_t)(featBase + l16) * EMB + (ks << 5) + (half << 4)];
                __builtin_prefetch(bp + 32, 0, 3);   // next k-step prefetch
                v16bf bfrag = *(const v16bf*)bp;
                acc0 = __builtin_amdgcn_wmma_f32_16x16x32_bf16(
                    false, a0.v, false, bfrag, (short)0, acc0, false, false);
                acc1 = __builtin_amdgcn_wmma_f32_16x16x32_bf16(
                    false, a1.v, false, bfrag, (short)0, acc1, false, false);
            }
        }
        const float bv = bias[featBase + l16];
        #pragma unroll
        for (int r = 0; r < 8; ++r) {
            // C layout: token = rt*16 + r + 8*half, feature col = featBase + l16
            int row = r + (half << 3);
            sh_qkv[row * Q3 + featBase + l16]        = (__bf16)(acc0[r] + bv);
            sh_qkv[(16 + row) * Q3 + featBase + l16] = (__bf16)(acc1[r] + bv);
        }
    }
    __syncthreads();

    // ---- phase 2: per-token attention over heads (2 tokens per wave) ----
    for (int tt = 0; tt < 2; ++tt) {
        const int t = wave * 2 + tt;
        const __bf16* qb = &sh_qkv[t * Q3];          // q: [head][64]
        const __bf16* kb = qb + EMB;                 // k
        const __bf16* vb = qb + 2 * EMB;             // v

        // scores(16x16) = Q @ K^T over K=64 (2 WMMA k-steps)
        v8f sacc = {};
        #pragma unroll
        for (int ks = 0; ks < 2; ++ks) {
            V16U a;
            const __bf16* ap = &qb[l16 * HD + (ks << 5) + (half << 4)];
            a.h[0] = *(const v8bf*)(ap);
            a.h[1] = *(const v8bf*)(ap + 8);
            const __bf16* bp = &kb[l16 * HD + (ks << 5) + (half << 4)];
            v16bf bfrag = *(const v16bf*)bp;
            sacc = __builtin_amdgcn_wmma_f32_16x16x32_bf16(
                false, a.v, false, bfrag, (short)0, sacc, false, false);
        }

        // row softmax: row i = r + 8*half lives in one 16-lane half;
        // reduce across j with xor-shuffles confined to the half.
        __bf16* at = &sh_at[wave * 16 * 32];
        #pragma unroll
        for (int e = 0; e < 8; ++e)                   // zero K-pad region j=16..31
            at[l16 * 32 + 16 + (half << 3) + e] = (__bf16)0.f;
        #pragma unroll
        for (int r = 0; r < 8; ++r) {
            float s = sacc[r] * 0.03125f;             // 1/sqrt(1024)
            float m = s;
            for (int msk = 1; msk < 16; msk <<= 1)
                m = fmaxf(m, __shfl_xor(m, msk, 32));
            float e = __expf(s - m);
            float sum = e;
            for (int msk = 1; msk < 16; msk <<= 1)
                sum += __shfl_xor(sum, msk, 32);
            at[(r + (half << 3)) * 32 + l16] = (__bf16)(e / sum);
        }

        // out(16x64) = attn(16x16, K padded to 32) @ V(16x64): 4 N-tiles
        #pragma unroll
        for (int nt = 0; nt < 4; ++nt) {
            V16U a;
            const __bf16* ap = &at[l16 * 32 + (half << 4)];
            a.h[0] = *(const v8bf*)(ap);
            a.h[1] = *(const v8bf*)(ap + 8);
            V16U bf_;
            #pragma unroll
            for (int e = 0; e < 16; ++e) {
                int j = e + (half << 4);              // half1 = pad rows -> 0
                bf_.v[e] = (j < NH) ? vb[j * HD + (nt << 4) + l16] : (__bf16)0.f;
            }
            v8f oacc = {};
            oacc = __builtin_amdgcn_wmma_f32_16x16x32_bf16(
                false, a.v, false, bf_.v, (short)0, oacc, false, false);
            #pragma unroll
            for (int r = 0; r < 8; ++r) {
                // reference layout: out flat index = head*(N*64) + token*64 + d
                int i = r + (half << 3);
                int d = (nt << 4) + l16;
                out[(size_t)i * (size_t)N_TOK * HD + (size_t)(tok0 + t) * HD + d] = oacc[r];
            }
        }
    }
}

extern "C" void kernel_launch(void* const* d_in, const int* in_sizes, int n_in,
                              void* d_out, int out_size, void* d_ws, size_t ws_size,
                              hipStream_t stream) {
    const float* h  = (const float*)d_in[0];
    const float* W  = (const float*)d_in[1];
    const float* b  = (const float*)d_in[2];
    float* out      = (float*)d_out;
    __bf16* Wb      = (__bf16*)d_ws;                 // 3072*1024*2 = 6 MB scratch

    // W f32 -> bf16 (stays L2-resident for all blocks of the fused kernel)
    cvt_w_bf16<<<(Q3 * EMB) / (256 * 4), 256, 0, stream>>>(W, Wb);

    const size_t smem = 65536 + 196608 + 16384;      // 272 KB dynamic LDS
    mha_fused<<<N_TOK / TM, NTHR, smem, stream>>>(h, Wb, b, out);
}
